// QAttn_30717606101509
// MI455X (gfx1250) — 16-bit and 8-bit matrix operand order repaired and checked by reading, compiled, not run on hardware
//
#include <hip/hip_runtime.h>
#include <hip/hip_fp16.h>
#include <stdint.h>

typedef __attribute__((ext_vector_type(8)))  int       i32x8;
typedef __attribute__((ext_vector_type(8)))  float     f32x8;
typedef __attribute__((ext_vector_type(16))) _Float16  f16x16;
typedef __attribute__((ext_vector_type(8)))  _Float16  f16x8;

#define QMAXF 127.0f
#define NSHIFT 30

__device__ __forceinline__ float qscale(const unsigned int* amax) {
  return fmaxf(__uint_as_float(*amax), 1e-8f) / QMAXF;
}
__device__ __forceinline__ float qclip(float x) {
  return fminf(fmaxf(x, -128.0f), QMAXF);
}

__device__ __forceinline__ f16x16 load_a16(const _Float16* rowbase, int h) {
  f16x8 lo = *(const f16x8*)(rowbase + 16 * h);
  f16x8 hi = *(const f16x8*)(rowbase + 16 * h + 8);
  return __builtin_shufflevector(lo, hi, 0,1,2,3,4,5,6,7,8,9,10,11,12,13,14,15);
}

__device__ __forceinline__ i32x8 load_b8(const signed char* p) {
  const int4* q = (const int4*)p;
  int4 b0 = q[0], b1 = q[2];
  i32x8 r;
  r[0]=b0.x; r[1]=b0.y; r[2]=b0.z; r[3]=b0.w;
  r[4]=b1.x; r[5]=b1.y; r[6]=b1.z; r[7]=b1.w;
  return r;
}

__device__ __forceinline__ void async_copy_b128(uint32_t lds_off, const void* gaddr) {
  asm volatile("global_load_async_to_lds_b128 %0, %1, off"
               :: "v"(lds_off), "v"(gaddr) : "memory");
}
__device__ __forceinline__ void wait_asynccnt0() {
  asm volatile("s_wait_asynccnt 0x0" ::: "memory");
}

__global__ void k_init(unsigned int* amax) {
  if (threadIdx.x < 4) amax[threadIdx.x] = 0u;
}

__global__ __launch_bounds__(256) void k_amax(const float* __restrict__ x,
                                              long long n, unsigned int* dst) {
  __shared__ float sb[256];
  float m = 0.f;
  for (long long i = (long long)blockIdx.x * blockDim.x + threadIdx.x; i < n;
       i += (long long)gridDim.x * blockDim.x)
    m = fmaxf(m, fabsf(x[i]));
  sb[threadIdx.x] = m; __syncthreads();
  for (int s = 128; s > 0; s >>= 1) {
    if (threadIdx.x < s) sb[threadIdx.x] = fmaxf(sb[threadIdx.x], sb[threadIdx.x + s]);
    __syncthreads();
  }
  if (threadIdx.x == 0) atomicMax(dst, __float_as_uint(sb[0]));
}

__global__ __launch_bounds__(256) void k_quant_rows(const float* __restrict__ w, int K,
                                                    signed char* __restrict__ wq,
                                                    float* __restrict__ srow) {
  __shared__ float sb[256];
  int row = blockIdx.x;
  const float* wr = w + (size_t)row * K;
  float m = 0.f;
  for (int i = threadIdx.x; i < K; i += 256) m = fmaxf(m, fabsf(wr[i]));
  sb[threadIdx.x] = m; __syncthreads();
  for (int s = 128; s > 0; s >>= 1) {
    if (threadIdx.x < s) sb[threadIdx.x] = fmaxf(sb[threadIdx.x], sb[threadIdx.x + s]);
    __syncthreads();
  }
  float sc = fmaxf(sb[0], 1e-8f) / QMAXF;
  if (threadIdx.x == 0) srow[row] = sc;
  for (int i = threadIdx.x; i < K; i += 256)
    wq[(size_t)row * K + i] = (signed char)(int)qclip(rintf(wr[i] / sc));
}

__global__ __launch_bounds__(256) void k_quant_tensor(const float* __restrict__ x, long long n,
                                                      const unsigned int* __restrict__ amax,
                                                      signed char* __restrict__ xq) {
  float sc = qscale(amax);
  for (long long i = (long long)blockIdx.x * blockDim.x + threadIdx.x; i < n;
       i += (long long)gridDim.x * blockDim.x)
    xq[i] = (signed char)(int)qclip(rintf(x[i] / sc));
}

__global__ __launch_bounds__(256) void k_gemm_i8(
    const signed char* __restrict__ A, const signed char* __restrict__ W,
    const unsigned int* __restrict__ aAmax, const float* __restrict__ sW,
    const float* __restrict__ bias, int M, int Nn, int K,
    _Float16* __restrict__ outH, float* __restrict__ outF,
    unsigned int* __restrict__ oAmax) {
  int lane = threadIdx.x & 31, wv = threadIdx.x >> 5;
  int ngroups = Nn >> 6;
  int total = (M >> 4) * ngroups;
  int wt = blockIdx.x * 8 + wv;
  if (wt >= total) return;
  int mt = wt / ngroups, ng = wt - mt * ngroups;
  int col = lane & 15, h = lane >> 4;
  float sa = qscale(aAmax);

  const signed char* Ab  = A + (size_t)(mt * 16 + col) * K + 16 * h;
  const signed char* Wb0 = W + (size_t)(ng * 64 +  0 + col) * K + 16 * h;
  const signed char* Wb1 = W + (size_t)(ng * 64 + 16 + col) * K + 16 * h;
  const signed char* Wb2 = W + (size_t)(ng * 64 + 32 + col) * K + 16 * h;
  const signed char* Wb3 = W + (size_t)(ng * 64 + 48 + col) * K + 16 * h;
  i32x8 acc0 = {0}, acc1 = {0}, acc2 = {0}, acc3 = {0};

  for (int kt = 0; kt < K; kt += 64) {
    const int* pa = (const int*)(Ab + kt);
    i32x8 a;
    a[0]=pa[0];  a[1]=pa[1];  a[2]=pa[2];  a[3]=pa[3];
    a[4]=pa[8];  a[5]=pa[9];  a[6]=pa[10]; a[7]=pa[11];
    i32x8 b0 = load_b8(Wb0 + kt);
    i32x8 b1 = load_b8(Wb1 + kt);
    i32x8 b2 = load_b8(Wb2 + kt);
    i32x8 b3 = load_b8(Wb3 + kt);
    if (kt + 64 < K) {
      __builtin_prefetch(Ab  + kt + 256);
      __builtin_prefetch(Wb0 + kt + 256);
      __builtin_prefetch(Wb2 + kt + 256);
    }
    acc0 = __builtin_amdgcn_wmma_i32_16x16x64_iu8(true, a, true, b0, acc0, false, false);
    acc1 = __builtin_amdgcn_wmma_i32_16x16x64_iu8(true, a, true, b1, acc1, false, false);
    acc2 = __builtin_amdgcn_wmma_i32_16x16x64_iu8(true, a, true, b2, acc2, false, false);
    acc3 = __builtin_amdgcn_wmma_i32_16x16x64_iu8(true, a, true, b3, acc3, false, false);
  }

  i32x8 accs[4] = {acc0, acc1, acc2, acc3};
  float lmax = 0.f;
#pragma unroll
  for (int j = 0; j < 4; ++j) {
    int n = ng * 64 + j * 16 + col;
    float scale = sa * sW[n];
    float bsv = bias[n];
#pragma unroll
    for (int v = 0; v < 8; ++v) {
      int m = mt * 16 + 8 * h + v;
      float val = (float)accs[j][v] * scale + bsv;
      if (outH) {
        outH[(size_t)m * Nn + n] = (_Float16)val;
        lmax = fmaxf(lmax, fabsf(val));
      } else {
        outF[(size_t)m * Nn + n] = val;
      }
    }
  }
  if (oAmax) atomicMax(oAmax, __float_as_uint(lmax));
}

__global__ __launch_bounds__(256) void k_split(const _Float16* __restrict__ qkvh,
                                               const unsigned int* __restrict__ amax,
                                               _Float16* __restrict__ Qh,
                                               _Float16* __restrict__ Kh,
                                               _Float16* __restrict__ Vt) {
  long long idx = (long long)blockIdx.x * blockDim.x + threadIdx.x;
  if (idx >= 8388608LL) return;
  float sc = qscale(amax);
  int d = (int)(idx & 63), n = (int)((idx >> 6) & 1023);
  int h = (int)((idx >> 16) & 15), b = (int)(idx >> 20);
  size_t base = ((size_t)(b * 1024 + n)) * 3072 + h * 64 + d;
  float q0 = qclip(rintf((float)qkvh[base]        / sc)) * sc;
  float k0 = qclip(rintf((float)qkvh[base + 1024] / sc)) * sc;
  float v0 = qclip(rintf((float)qkvh[base + 2048] / sc)) * sc;
  Qh[idx] = (_Float16)(q0 * 0.125f);
  Kh[idx] = (_Float16)k0;
  Vt[((size_t)(b * 16 + h) * 64 + d) * 1024 + n] = (_Float16)v0;
}

__global__ __launch_bounds__(256) void k_amax_s(const _Float16* __restrict__ Qh,
                                                const _Float16* __restrict__ Kh,
                                                unsigned int* dst) {
  __shared__ float sb[256];
  int lane = threadIdx.x & 31, wv = threadIdx.x >> 5;
  int wt = blockIdx.x * 8 + wv;
  int bh = wt >> 12, rem = wt & 4095;
  int mt = rem >> 6, nt = rem & 63;
  int col = lane & 15, h = lane >> 4;
  const _Float16* Qb = Qh + (size_t)bh * 65536;
  const _Float16* Kb = Kh + (size_t)bh * 65536;
  f16x16 a0 = load_a16(Qb + (size_t)(mt * 16 + col) * 64 +  0, h);
  f16x16 a1 = load_a16(Qb + (size_t)(mt * 16 + col) * 64 + 32, h);
  f16x16 b0 = *(const f16x16*)(Kb + (size_t)(nt * 16 + col) * 64 +  0 + 16 * h);
  f16x16 b1 = *(const f16x16*)(Kb + (size_t)(nt * 16 + col) * 64 + 32 + 16 * h);
  f32x8 acc = {};
  acc = __builtin_amdgcn_wmma_f32_16x16x32_f16(false, a0, false, b0, (short)0, acc, false, false);
  acc = __builtin_amdgcn_wmma_f32_16x16x32_f16(false, a1, false, b1, (short)0, acc, false, false);
  float m = 0.f;
#pragma unroll
  for (int v = 0; v < 8; ++v) m = fmaxf(m, fabsf(acc[v]));
  sb[threadIdx.x] = m; __syncthreads();
  for (int s = 128; s > 0; s >>= 1) {
    if (threadIdx.x < s) sb[threadIdx.x] = fmaxf(sb[threadIdx.x], sb[threadIdx.x + s]);
    __syncthreads();
  }
  if (threadIdx.x == 0) atomicMax(dst, __float_as_uint(sb[0]));
}

__global__ __launch_bounds__(256) void k_attn(const _Float16* __restrict__ Qh,
                                              const _Float16* __restrict__ Kh,
                                              const _Float16* __restrict__ Vt,
                                              const unsigned int* __restrict__ sAmax,
                                              float* __restrict__ obuf,
                                              unsigned int* __restrict__ oAmax) {
  extern __shared__ char smem[];
  float*    S   = (float*)smem;
  _Float16* P   = (_Float16*)(smem + 65536);
  float*    red = (float*)(smem + 65536 + 33024);
  _Float16* Qt  = (_Float16*)(smem + 65536 + 33024 + 2048);
  float*    OP  = S;

  int lane = threadIdx.x & 31, wv = threadIdx.x >> 5;
  int col = lane & 15, h = lane >> 4;
  int bh = blockIdx.x >> 6, mt = blockIdx.x & 63;
  const _Float16* Qb = Qh + (size_t)bh * 65536;
  const _Float16* Kb = Kh + (size_t)bh * 65536;
  const _Float16* Vb = Vt + (size_t)bh * 65536;

  if (threadIdx.x < 128) {
    int t = threadIdx.x;
    const void* g = (const void*)(Qb + (size_t)(mt * 16 + (t >> 3)) * 64 + (t & 7) * 8);
    uint32_t lds_off = (uint32_t)(uintptr_t)(Qt + t * 8);
    async_copy_b128(lds_off, g);
  }
  wait_asynccnt0();
  __syncthreads();

  f16x16 qa0 = load_a16(Qt + (size_t)col * 64 +  0, h);
  f16x16 qa1 = load_a16(Qt + (size_t)col * 64 + 32, h);
  for (int t = wv; t < 64; t += 8) {
    f16x16 b0 = *(const f16x16*)(Kb + (size_t)(t * 16 + col) * 64 +  0 + 16 * h);
    f16x16 b1 = *(const f16x16*)(Kb + (size_t)(t * 16 + col) * 64 + 32 + 16 * h);
    f32x8 acc = {};
    acc = __builtin_amdgcn_wmma_f32_16x16x32_f16(false, qa0, false, b0, (short)0, acc, false, false);
    acc = __builtin_amdgcn_wmma_f32_16x16x32_f16(false, qa1, false, b1, (short)0, acc, false, false);
#pragma unroll
    for (int v = 0; v < 8; ++v) S[(v + 8 * h) * 1024 + t * 16 + col] = acc[v];
  }
  __syncthreads();

  float s    = qscale(sAmax);
  float x0   = floorf(-0.6931471805599453f / s);
  float bint = floorf(1.353f / s);
  float cint = floorf(0.344f / (0.35815147f * s * s));
  int r  = threadIdx.x >> 4;
  int ts = threadIdx.x & 15;

  float qmax = -3.0e38f;
  for (int i = ts; i < 1024; i += 16) {
    float qi = qclip(rintf(S[r * 1024 + i] / s));
    S[r * 1024 + i] = qi;
    qmax = fmaxf(qmax, qi);
  }
  red[r * 16 + ts] = qmax;
  __syncthreads();
  float rmax = red[r * 16];
#pragma unroll
  for (int j = 1; j < 16; ++j) rmax = fmaxf(rmax, red[r * 16 + j]);
  __syncthreads();

  float qfloor = (float)NSHIFT * x0;
  float esum = 0.f;
  for (int i = ts; i < 1024; i += 16) {
    float qv = fmaxf(S[r * 1024 + i] - rmax, qfloor);
    float z  = floorf(qv / x0);
    float rr = qv - x0 * z;
    float poly = (rr + bint) * rr + cint;
    float e = fmaxf(floorf(ldexpf(poly, NSHIFT - (int)z)), 0.f);
    S[r * 1024 + i] = e;
    esum += e;
  }
  red[r * 16 + ts] = esum;
  __syncthreads();
  float rsum = 0.f;
#pragma unroll
  for (int j = 0; j < 16; ++j) rsum += red[r * 16 + j];
  float inv = 1.0f / rsum;
  for (int i = ts; i < 1024; i += 16)
    P[r * 1032 + i] = (_Float16)(S[r * 1024 + i] * inv);
  __syncthreads();

  f32x8 acc[4] = {{}, {}, {}, {}};
  int k0 = wv * 128;
  for (int kt = k0; kt < k0 + 128; kt += 32) {
    f16x16 a = load_a16(P + (size_t)col * 1032 + kt, h);
#pragma unroll
    for (int j = 0; j < 4; ++j) {
      f16x16 b = *(const f16x16*)(Vb + (size_t)(j * 16 + col) * 1024 + kt + 16 * h);
      acc[j] = __builtin_amdgcn_wmma_f32_16x16x32_f16(false, a, false, b, (short)0, acc[j], false, false);
    }
  }
#pragma unroll
  for (int j = 0; j < 4; ++j)
#pragma unroll
    for (int v = 0; v < 8; ++v)
      OP[wv * 1024 + (v + 8 * h) * 64 + j * 16 + col] = acc[j][v];
  __syncthreads();

  int b_ = bh >> 4, hh = bh & 15;
  float lmax = 0.f;
  for (int e = threadIdx.x; e < 1024; e += 256) {
    int m = e >> 6, d = e & 63;
    float sum = 0.f;
#pragma unroll
    for (int w = 0; w < 8; ++w) sum += OP[w * 1024 + m * 64 + d];
    obuf[((size_t)b_ * 1024 + mt * 16 + m) * 1024 + hh * 64 + d] = sum;
    lmax = fmaxf(lmax, fabsf(sum));
  }
  atomicMax(oAmax, __float_as_uint(lmax));
}

extern "C" void kernel_launch(void* const* d_in, const int* in_sizes, int n_in,
                              void* d_out, int out_size, void* d_ws, size_t ws_size,
                              hipStream_t stream) {
  const float* x      = (const float*)d_in[0];
  const float* w_qkv  = (const float*)d_in[1];
  const float* b_qkv  = (const float*)d_in[2];
  const float* w_proj = (const float*)d_in[3];
  const float* b_proj = (const float*)d_in[4];
  float* out = (float*)d_out;

  char* ws = (char*)d_ws;
  unsigned int* amax = (unsigned int*)ws;
  float* s_wqkv  = (float*)(ws + 256);
  float* s_wproj = (float*)(ws + 256 + 3072 * 4);
  size_t off = 32768;
  signed char* xq     = (signed char*)(ws + off); off += (size_t)8192 * 1024;
  signed char* wqkvq  = (signed char*)(ws + off); off += (size_t)3072 * 1024;
  signed char* wprojq = (signed char*)(ws + off); off += (size_t)1024 * 1024;
  _Float16* qkvh = (_Float16*)(ws + off); off += (size_t)8192 * 3072 * 2;
  _Float16* Qh   = (_Float16*)(ws + off); off += (size_t)8388608 * 2;
  _Float16* Kh   = (_Float16*)(ws + off); off += (size_t)8388608 * 2;
  _Float16* Vt   = (_Float16*)(ws + off); off += (size_t)8388608 * 2;
  float* obuf    = (float*)(ws + off);    off += (size_t)8388608 * 4;
  signed char* oq = (signed char*)(ws + off); off += (size_t)8388608;
  (void)in_sizes; (void)n_in; (void)out_size; (void)ws_size;

  k_init<<<1, 32, 0, stream>>>(amax);
  k_amax<<<512, 256, 0, stream>>>(x, 8388608LL, amax + 0);
  k_quant_rows<<<3072, 256, 0, stream>>>(w_qkv, 1024, wqkvq, s_wqkv);
  k_quant_rows<<<1024, 256, 0, stream>>>(w_proj, 1024, wprojq, s_wproj);
  k_quant_tensor<<<2048, 256, 0, stream>>>(x, 8388608LL, amax + 0, xq);

  k_gemm_i8<<<3072, 256, 0, stream>>>(xq, wqkvq, amax + 0, s_wqkv, b_qkv,
                                      8192, 3072, 1024, qkvh, nullptr, amax + 1);
  k_split<<<32768, 256, 0, stream>>>(qkvh, amax + 1, Qh, Kh, Vt);

  k_amax_s<<<65536, 256, 0, stream>>>(Qh, Kh, amax + 2);
  size_t shbytes = 65536 + 33024 + 2048 + 2048;
  k_attn<<<8192, 256, shbytes, stream>>>(Qh, Kh, Vt, amax + 2, obuf, amax + 3);

  k_quant_tensor<<<2048, 256, 0, stream>>>(obuf, 8388608LL, amax + 3, oq);
  k_gemm_i8<<<1024, 256, 0, stream>>>(oq, wprojq, amax + 3, s_wproj, b_proj,
                                      8192, 1024, 1024, nullptr, out, nullptr);
}
